// NTXEntLoss_3332894622322
// MI455X (gfx1250) — compile-verified, matrix operand order checked by reading, not run on hardware
//
#include <hip/hip_runtime.h>
#include <hip/hip_bf16.h>
#include <math.h>

typedef __attribute__((ext_vector_type(16))) _Float16 v16h;
typedef __attribute__((ext_vector_type(8)))  _Float16 v8h;
typedef __attribute__((ext_vector_type(8)))  float    v8f;

#define B_ 512
#define D_ 256
#define H_ 32
#define W_ 32

// workspace layout (float element offsets)
#define OFF_D1   0         // 131072 floats: d1 (f32)
#define OFF_D2   131072    // 131072 floats: d2 (f32)
#define OFF_H16  262144    // 262144 halfs (=131072 floats): d1h then d2h
#define OFF_N1   393216    // 512
#define OFF_N2   393728    // 512
#define OFF_SIM  394240    // 262144: sim/T (f32)
#define OFF_RMAX 656384    // 512: max over j of sim[i,:]
#define OFF_CMAX 656896    // 512: max over i of sim[:,j]
#define OFF_RDEN 657408    // 512
#define OFF_CDEN 657920    // 512

// ---------------------------------------------------------------------------
// Kernel 1: box-masked average pooling.  One wave per (b,d) plane; lane = x.
// Only rows with y-center inside the box are loaded (full 128B coalesced row).
// ---------------------------------------------------------------------------
__global__ __launch_bounds__(256)
void pool_kernel(const float* __restrict__ f1, const float* __restrict__ f2,
                 const float* __restrict__ bb1, const float* __restrict__ bb2,
                 float* __restrict__ ws) {
  const int wave = threadIdx.x >> 5;
  const int lane = threadIdx.x & 31;
  const int d = blockIdx.x * 8 + wave;
  const int b = blockIdx.y;
  const int t = blockIdx.z;

  const float* F  = t ? f2  : f1;
  const float* Bx = t ? bb2 : bb1;
  float*     desc = ws + (t ? OFF_D2 : OFF_D1);
  _Float16* desch = ((_Float16*)(ws + OFF_H16)) + (t ? (B_ * D_) : 0);

  const float x0 = Bx[b * 4 + 0];
  const float y0 = Bx[b * 4 + 1];
  const float x1 = Bx[b * 4 + 2];
  const float y1 = Bx[b * 4 + 3];

  const float xc = ((float)lane + 0.5f) * (1.0f / (float)W_);
  const bool  mx = (xc >= x0) && (xc <= x1);

  const float* plane = F + (size_t)(b * D_ + d) * (H_ * W_);

  float acc = 0.0f;
  float cnt = 0.0f;
  for (int y = 0; y < H_; ++y) {
    const float yc = ((float)y + 0.5f) * (1.0f / (float)H_);
    if (yc >= y0 && yc <= y1) {           // scalar (wave-uniform) branch
      float v = plane[y * W_ + lane];     // 128B coalesced row load
      if (mx) { acc += v; cnt += 1.0f; }
    }
  }
  // wave32 reduction
  #pragma unroll
  for (int off = 16; off > 0; off >>= 1) {
    acc += __shfl_xor(acc, off, 32);
    cnt += __shfl_xor(cnt, off, 32);
  }
  if (lane == 0) {
    const float denom = fmaxf(cnt, 1.0f);
    const float r = acc / denom;
    desc[b * D_ + d]  = r;
    desch[b * D_ + d] = (_Float16)r;
  }
}

// ---------------------------------------------------------------------------
// Kernel 2: per-descriptor L2 norms.
// ---------------------------------------------------------------------------
__global__ __launch_bounds__(256)
void norm_kernel(float* __restrict__ ws) {
  __shared__ float s[256];
  const int b = blockIdx.x;
  const int t = blockIdx.y;
  const float* desc = ws + (t ? OFF_D2 : OFF_D1);
  const float v = desc[b * D_ + threadIdx.x];
  s[threadIdx.x] = v * v;
  __syncthreads();
  for (int st = 128; st > 0; st >>= 1) {
    if (threadIdx.x < st) s[threadIdx.x] += s[threadIdx.x + st];
    __syncthreads();
  }
  if (threadIdx.x == 0) (ws + (t ? OFF_N2 : OFF_N1))[b] = sqrtf(s[0]);
}

// ---------------------------------------------------------------------------
// Kernel 3: sim/T = (d1 @ d2^T) / max(n1*n2, 1e-8) / 0.1 via WMMA f16.
// Each wave computes one 16x16 tile; K=256 => 8x v_wmma_f32_16x16x32_f16.
// A-frag layout (ISA 7.12.2): lane L (row = L&15, half = L>>4) holds
// K = [8*half .. 8*half+7] and [16+8*half .. 16+8*half+7] per 32-wide K-step.
// B = d2^T, so B column n is row n of d2 with the identical per-lane packing.
// ---------------------------------------------------------------------------
__device__ inline v16h load_frag(const _Float16* __restrict__ p, int half) {
  const v8h lo = *(const v8h*)(p + 8 * half);
  const v8h hi = *(const v8h*)(p + 16 + 8 * half);
  v16h f;
  #pragma unroll
  for (int e = 0; e < 8; ++e) { f[e] = lo[e]; f[e + 8] = hi[e]; }
  return f;
}

__global__ __launch_bounds__(256)
void gemm_kernel(float* __restrict__ ws) {
  const _Float16* hb  = (const _Float16*)(ws + OFF_H16);
  const _Float16* d1h = hb;
  const _Float16* d2h = hb + B_ * D_;
  const float* n1 = ws + OFF_N1;
  const float* n2 = ws + OFF_N2;
  float* sim = ws + OFF_SIM;

  const int tile = blockIdx.x * 8 + (threadIdx.x >> 5);
  const int tm = tile >> 5;          // row tile 0..31
  const int tn = tile & 31;          // col tile 0..31
  const int lane = threadIdx.x & 31;
  const int r    = lane & 15;
  const int half = lane >> 4;

  const _Float16* arow = d1h + (size_t)(tm * 16 + r) * D_;
  const _Float16* brow = d2h + (size_t)(tn * 16 + r) * D_;

  v8f c = {};
  #pragma unroll
  for (int k0 = 0; k0 < D_; k0 += 32) {
    v16h a = load_frag(arow + k0, half);
    v16h b = load_frag(brow + k0, half);
    c = __builtin_amdgcn_wmma_f32_16x16x32_f16(
        /*neg_a=*/false, a, /*neg_b=*/false, b,
        /*c_mod=*/(short)0, c, /*reuse_a=*/false, /*reuse_b=*/false);
  }

  // C/D layout: VGPR j -> M = j + 8*half, N = lane & 15
  #pragma unroll
  for (int j = 0; j < 8; ++j) {
    const int m = tm * 16 + half * 8 + j;
    const int n = tn * 16 + r;
    const float den = fmaxf(n1[m] * n2[n], 1e-8f);
    sim[(size_t)m * B_ + n] = (c[j] / den) / 0.1f;   // already divided by T
  }
}

// ---------------------------------------------------------------------------
// Kernel 4: rmax[i] = max_j sim[i,j] (y==0); cmax[j] = max_i sim[i,j] (y==1)
// ---------------------------------------------------------------------------
__global__ __launch_bounds__(256)
void max_kernel(float* __restrict__ ws) {
  __shared__ float s[256];
  const float* sim = ws + OFF_SIM;
  const int k = blockIdx.x;
  float m;
  if (blockIdx.y == 0) {
    m = fmaxf(sim[(size_t)k * B_ + threadIdx.x],
              sim[(size_t)k * B_ + threadIdx.x + 256]);
  } else {
    m = fmaxf(sim[(size_t)threadIdx.x * B_ + k],
              sim[(size_t)(threadIdx.x + 256) * B_ + k]);
  }
  s[threadIdx.x] = m;
  __syncthreads();
  for (int st = 128; st > 0; st >>= 1) {
    if (threadIdx.x < st) s[threadIdx.x] = fmaxf(s[threadIdx.x], s[threadIdx.x + st]);
    __syncthreads();
  }
  if (threadIdx.x == 0) (ws + (blockIdx.y ? OFF_CMAX : OFF_RMAX))[k] = s[0];
}

// ---------------------------------------------------------------------------
// Kernel 5 (matches reference broadcasting exactly):
//   row_den[k] = sum_j exp(sim[k,j] - cmax[j])    (per-j max inside sum)
//   col_den[k] = sum_i exp(sim[i,k] - rmax[k])    (fixed max rmax[k])
// (diagonal terms equal the respective num terms, so full sums are correct)
// ---------------------------------------------------------------------------
__global__ __launch_bounds__(256)
void den_kernel(float* __restrict__ ws) {
  __shared__ float s[256];
  const float* sim  = ws + OFF_SIM;
  const float* rmax = ws + OFF_RMAX;
  const float* cmax = ws + OFF_CMAX;
  const int k = blockIdx.x;
  float a;
  if (blockIdx.y == 0) {
    const int j0 = threadIdx.x, j1 = threadIdx.x + 256;
    a = expf(sim[(size_t)k * B_ + j0] - cmax[j0]) +
        expf(sim[(size_t)k * B_ + j1] - cmax[j1]);
  } else {
    const int i0 = threadIdx.x, i1 = threadIdx.x + 256;
    const float mk = rmax[k];
    a = expf(sim[(size_t)i0 * B_ + k] - mk) +
        expf(sim[(size_t)i1 * B_ + k] - mk);
  }
  s[threadIdx.x] = a;
  __syncthreads();
  for (int st = 128; st > 0; st >>= 1) {
    if (threadIdx.x < st) s[threadIdx.x] += s[threadIdx.x + st];
    __syncthreads();
  }
  if (threadIdx.x == 0) (ws + (blockIdx.y ? OFF_CDEN : OFF_RDEN))[k] = s[0];
}

// ---------------------------------------------------------------------------
// Kernel 6: final loss = sum_k -(row_log + col_log) / (2B)
// ---------------------------------------------------------------------------
__global__ __launch_bounds__(256)
void loss_kernel(const float* __restrict__ ws, float* __restrict__ out) {
  __shared__ float s[256];
  const float* sim  = ws + OFF_SIM;
  const float* rmax = ws + OFF_RMAX;
  const float* cmax = ws + OFF_CMAX;
  const float* rden = ws + OFF_RDEN;
  const float* cden = ws + OFF_CDEN;

  float acc = 0.0f;
  for (int k = threadIdx.x; k < B_; k += 256) {
    const float diag = sim[(size_t)k * B_ + k];
    const float rlog = logf(expf(diag - cmax[k]) / rden[k] + 1e-20f);
    const float clog = logf(expf(diag - rmax[k]) / cden[k] + 1e-20f);
    acc += -(rlog + clog);
  }
  s[threadIdx.x] = acc;
  __syncthreads();
  for (int st = 128; st > 0; st >>= 1) {
    if (threadIdx.x < st) s[threadIdx.x] += s[threadIdx.x + st];
    __syncthreads();
  }
  if (threadIdx.x == 0) out[0] = s[0] / (2.0f * (float)B_);
}

// ---------------------------------------------------------------------------
extern "C" void kernel_launch(void* const* d_in, const int* in_sizes, int n_in,
                              void* d_out, int out_size, void* d_ws, size_t ws_size,
                              hipStream_t stream) {
  const float* f1  = (const float*)d_in[0];
  const float* f2  = (const float*)d_in[1];
  const float* bb1 = (const float*)d_in[2];
  const float* bb2 = (const float*)d_in[3];
  float* ws  = (float*)d_ws;
  float* out = (float*)d_out;

  // 1) pooling: grid (D/8, B, 2), one wave per (b,d) plane
  pool_kernel<<<dim3(D_ / 8, B_, 2), 256, 0, stream>>>(f1, f2, bb1, bb2, ws);
  // 2) norms
  norm_kernel<<<dim3(B_, 2), 256, 0, stream>>>(ws);
  // 3) WMMA GEMM: 1024 tiles of 16x16, 8 waves per block
  gemm_kernel<<<dim3(128), 256, 0, stream>>>(ws);
  // 4) row/col maxes
  max_kernel<<<dim3(B_, 2), 256, 0, stream>>>(ws);
  // 5) denominators
  den_kernel<<<dim3(B_, 2), 256, 0, stream>>>(ws);
  // 6) final scalar
  loss_kernel<<<dim3(1), 256, 0, stream>>>(ws, out);
}
